// MoEGate_13907104105110
// MI455X (gfx1250) — matrix operand order repaired and checked by reading, not compiled, not run on hardware
//
#include <hip/hip_runtime.h>
#include <hip/hip_bf16.h>

typedef __attribute__((ext_vector_type(16))) __bf16 v16bf;
typedef __attribute__((ext_vector_type(8)))  __bf16 v8bf;
typedef __attribute__((ext_vector_type(8)))  float  v8f;

#define TOKENS   32768
#define HIDDEN   5120
#define NEXP     160
#define NGROUP   8
#define EPG      20      // experts per group
#define TOPG     3
#define TOPK     6
#define BM       64      // tokens per workgroup (4 waves x 16 tokens)
#define NT       10      // 160 / 16 expert tiles
#define LDS_STRIDE 161   // padded row stride (odd -> conflict-free columns)

// ---------------- W pre-conversion: fp32 -> bf16, once per launch ----------------
// 160*5120 = 819200 elements; 8 per thread -> 102400 threads -> 400 blocks x 256
__global__ __launch_bounds__(256)
void convert_w_kernel(const float* __restrict__ W, __bf16* __restrict__ Wb) {
    const size_t i = ((size_t)blockIdx.x * blockDim.x + threadIdx.x) * 8;
    float4 f0 = *reinterpret_cast<const float4*>(W + i);
    float4 f1 = *reinterpret_cast<const float4*>(W + i + 4);
    v8bf r;
    r[0] = (__bf16)f0.x; r[1] = (__bf16)f0.y; r[2] = (__bf16)f0.z; r[3] = (__bf16)f0.w;
    r[4] = (__bf16)f1.x; r[5] = (__bf16)f1.y; r[6] = (__bf16)f1.z; r[7] = (__bf16)f1.w;
    *reinterpret_cast<v8bf*>(Wb + i) = r;
}

__device__ __forceinline__ v16bf pack16_bf16(float4 f0, float4 f1, float4 f2, float4 f3) {
    v16bf r;
    r[0]  = (__bf16)f0.x; r[1]  = (__bf16)f0.y; r[2]  = (__bf16)f0.z; r[3]  = (__bf16)f0.w;
    r[4]  = (__bf16)f1.x; r[5]  = (__bf16)f1.y; r[6]  = (__bf16)f1.z; r[7]  = (__bf16)f1.w;
    r[8]  = (__bf16)f2.x; r[9]  = (__bf16)f2.y; r[10] = (__bf16)f2.z; r[11] = (__bf16)f2.w;
    r[12] = (__bf16)f3.x; r[13] = (__bf16)f3.y; r[14] = (__bf16)f3.z; r[15] = (__bf16)f3.w;
    return r;
}

// ---------------- Fused gating GEMM + routing ----------------
__global__ __launch_bounds__(128)
void moe_gate_kernel(const float* __restrict__ X,     // [TOKENS, HIDDEN] f32
                     const __bf16* __restrict__ Wb,   // [NEXP, HIDDEN] bf16 (pre-converted)
                     float* __restrict__ out) {       // [TOKENS, TOPK]  f32
    __shared__ float logits[BM * LDS_STRIDE];

    const int lane   = threadIdx.x & 31;
    const int wave   = threadIdx.x >> 5;
    const int lrow   = lane & 15;          // M (A) or N (B) index within tile
    const int hiHalf = (lane >> 4) & 1;    // which K half this lane supplies
    const int tok0   = blockIdx.x * BM + wave * 16;

    v8f acc[NT] = {};

    // A fragment: 16-bit A 16x32 layout -> low lanes K {0..7,16..23}, high lanes K {8..15,24..31}
    const float* aBase = X + (size_t)(tok0 + lrow) * HIDDEN + hiHalf * 8;

    for (int k0 = 0; k0 < HIDDEN; k0 += 32) {
        const float* ap = aBase + k0;
        float4 a0 = *reinterpret_cast<const float4*>(ap);
        float4 a1 = *reinterpret_cast<const float4*>(ap + 4);
        float4 a2 = *reinterpret_cast<const float4*>(ap + 16);
        float4 a3 = *reinterpret_cast<const float4*>(ap + 20);
        __builtin_prefetch(ap + 32, 0, 3);   // global_prefetch_b8: next K chunk of this token row
        v16bf A = pack16_bf16(a0, a1, a2, a3);

        #pragma unroll
        for (int tt = 0; tt < NT; ++tt) {
            // 16-bit B 32x16 layout: low lanes K {0..7,16..23}, high lanes K {8..15,24..31}; lane -> expert column.
            // Pre-converted bf16: 2 x 8 contiguous halfs = 2 x b128 loads, zero VALU.
            const __bf16* bp = Wb + (size_t)(tt * 16 + lrow) * HIDDEN + k0 + hiHalf * 8;
            union { v16bf v; v8bf h[2]; } B;
            B.h[0] = *reinterpret_cast<const v8bf*>(bp);
            B.h[1] = *reinterpret_cast<const v8bf*>(bp + 16);
            acc[tt] = __builtin_amdgcn_wmma_f32_16x16x32_bf16(
                false, A, false, B.v, (short)0, acc[tt], false, false);
        }
    }

    // C layout: VGPR r, lanes 0-15 -> (M=r, N=lane); lanes 16-31 -> (M=8+r, N=lane-16)
    #pragma unroll
    for (int tt = 0; tt < NT; ++tt) {
        #pragma unroll
        for (int r = 0; r < 8; ++r) {
            int m = wave * 16 + hiHalf * 8 + r;
            int e = tt * 16 + lrow;
            logits[m * LDS_STRIDE + e] = acc[tt][r];
        }
    }
    __syncthreads();

    // ---- Routing epilogue: one thread per token ----
    if (threadIdx.x < BM) {
        const float* row = &logits[threadIdx.x * LDS_STRIDE];
        const float NEG = -3.402823466e38f;

        // group maxes on logits (softmax is monotonic -> same ranking as on scores)
        float g[NGROUP];
        float mAll = NEG;
        #pragma unroll
        for (int gi = 0; gi < NGROUP; ++gi) {
            float mg = row[gi * EPG];
            for (int j = 1; j < EPG; ++j) mg = fmaxf(mg, row[gi * EPG + j]);
            g[gi] = mg;
            mAll = fmaxf(mAll, mg);
        }

        // full softmax denominator over all 160 experts
        float sum = 0.0f;
        for (int e = 0; e < NEXP; ++e) sum += __expf(row[e] - mAll);

        // top-3 groups (first-occurrence tie-break like lax.top_k)
        float gw[NGROUP];
        #pragma unroll
        for (int gi = 0; gi < NGROUP; ++gi) gw[gi] = g[gi];
        int sel[TOPG];
        for (int s = 0; s < TOPG; ++s) {
            int bi = 0; float bv = gw[0];
            for (int gi = 1; gi < NGROUP; ++gi)
                if (gw[gi] > bv) { bv = gw[gi]; bi = gi; }
            sel[s] = bi;
            gw[bi] = NEG;
        }

        // top-6 logits among the 3 selected groups (insertion into sorted array)
        float top[TOPK];
        #pragma unroll
        for (int j = 0; j < TOPK; ++j) top[j] = NEG;
        for (int s = 0; s < TOPG; ++s) {
            const float* gr = row + sel[s] * EPG;
            for (int j = 0; j < EPG; ++j) {
                float v = gr[j];
                if (v > top[TOPK - 1]) {
                    top[TOPK - 1] = v;
                    for (int q = TOPK - 1; q > 0; --q) {
                        if (top[q] > top[q - 1]) { float t = top[q - 1]; top[q - 1] = top[q]; top[q] = t; }
                        else break;
                    }
                }
            }
        }

        const float scale = 16.0f / sum;   // ROUTED_SCALING / softmax denom
        float* o = out + (size_t)(blockIdx.x * BM + threadIdx.x) * TOPK;
        #pragma unroll
        for (int j = 0; j < TOPK; ++j)
            o[j] = __expf(top[j] - mAll) * scale;
    }
}

extern "C" void kernel_launch(void* const* d_in, const int* in_sizes, int n_in,
                              void* d_out, int out_size, void* d_ws, size_t ws_size,
                              hipStream_t stream) {
    (void)in_sizes; (void)n_in; (void)ws_size; (void)out_size;
    const float* X = (const float*)d_in[0];   // hidden_states [32768, 5120] f32
    const float* W = (const float*)d_in[1];   // kernel        [160, 5120]  f32
    float* out = (float*)d_out;               // [32768, 6] f32
    __bf16* Wb = (__bf16*)d_ws;               // 160*5120*2 = 1.6 MB scratch

    // Stage 1: W fp32 -> bf16 (same stream => ordered before the GEMM; re-runs
    // every call so the launch is deterministic under graph capture/replay).
    convert_w_kernel<<<dim3((NEXP * HIDDEN) / (256 * 8)), dim3(256), 0, stream>>>(W, Wb);

    // Stage 2: fused gating GEMM (bf16 WMMA) + softmax/group-top-k routing.
    moe_gate_kernel<<<dim3(TOKENS / BM), dim3(128), 0, stream>>>(X, Wb, out);
}
